// SelfAttention_7619271983456
// MI455X (gfx1250) — compile-verified, matrix operand order checked by reading, not run on hardware
//
#include <hip/hip_runtime.h>
#include <hip/hip_bf16.h>
#include <math.h>

// ---------------- problem constants ----------------
constexpr int Bn  = 4;      // batch
constexpr int Cd  = 256;    // channels
constexpr int Cq  = 32;     // query/key dim
constexpr int Nd  = 4096;   // H*W
constexpr float CLAMP = 5.0f;

// ---------------- vector types ----------------
typedef __attribute__((ext_vector_type(8)))  float           v8f;
typedef __attribute__((ext_vector_type(16))) __bf16          v16bf;
typedef __attribute__((ext_vector_type(8)))  unsigned short  v8u;
typedef __attribute__((ext_vector_type(16))) unsigned short  v16u;
typedef __attribute__((ext_vector_type(4)))  int             v4i;

typedef __attribute__((address_space(1))) v4i* as1_v4i;
typedef __attribute__((address_space(3))) v4i* as3_v4i;

// Native f32 -> bf16 (let the backend pick v_cvt_*_bf16 instead of bit-twiddle VALU)
static __device__ inline unsigned short f2bf(float f) {
  union { __bf16 h; unsigned short u; } x;
  x.h = (__bf16)f;
  return x.u;
}

static __device__ inline v16bf as_bf(v16u u) {
  union { v16u u; v16bf b; } x; x.u = u; return x.b;
}

static __device__ inline v8f zero8() {
  v8f z;
  #pragma unroll
  for (int i = 0; i < 8; ++i) z[i] = 0.0f;
  return z;
}

// Load a WMMA 16-bit operand from a contiguous 32-element bf16 row.
// Lane mapping (16-bit A/B layout): half = lane>>4 selects K-subsets
// {h*8..h*8+7} and {h*8+16..h*8+23}; each is one 16-byte load.
static __device__ inline v16bf load_row_operand(const unsigned short* row, int half) {
  const v8u lo = *(const v8u*)(row + half * 8);
  const v8u hi = *(const v8u*)(row + half * 8 + 16);
  v16u u;
  #pragma unroll
  for (int i = 0; i < 8; ++i) { u[i] = lo[i]; u[8 + i] = hi[i]; }
  return as_bf(u);
}

static __device__ inline v8f wmma_bf16(v16bf a, v16bf b, v8f c) {
  return __builtin_amdgcn_wmma_f32_16x16x32_bf16(
      /*neg_a=*/false, a, /*neg_b=*/false, b,
      /*c_mod=*/(short)0, c, /*reuse_a=*/false, /*reuse_b=*/false);
}

// ---- CDNA5 async global->LDS copy (ASYNCcnt path), with safe fallback ----
static __device__ inline void async_copy_b128(const unsigned short* g, unsigned short* l) {
#if __has_builtin(__builtin_amdgcn_global_load_async_to_lds_b128)
  // AS1 pointer: generic global pointer is value-identical.
  // AS3 pointer: low 32 bits of a generic LDS pointer are the LDS offset.
  __builtin_amdgcn_global_load_async_to_lds_b128(
      (as1_v4i)(unsigned long long)g,
      (as3_v4i)(unsigned int)(unsigned long long)l,
      0, 0);
#else
  *(v8u*)l = *(const v8u*)g;
#endif
}

static __device__ inline void wait_async() {
#if __has_builtin(__builtin_amdgcn_global_load_async_to_lds_b128)
# if __has_builtin(__builtin_amdgcn_s_wait_asynccnt)
  __builtin_amdgcn_s_wait_asynccnt(0);
# else
  asm volatile("s_wait_asynccnt 0x0" ::: "memory");
# endif
#endif
}

// ============================================================
// Kernel 0: weights f32 -> bf16 (one-shot, removes all conversion
// VALU from the projection inner loops)
// ============================================================
__global__ __launch_bounds__(256)
void k_cvt_weights(const float* __restrict__ Wq, const float* __restrict__ Wk,
                   const float* __restrict__ Wv,
                   unsigned short* __restrict__ Wqb, unsigned short* __restrict__ Wkb,
                   unsigned short* __restrict__ Wvb) {
  const int i = blockIdx.x * 256 + threadIdx.x;
  if (i < Cq * Cd) { Wqb[i] = f2bf(Wq[i]); Wkb[i] = f2bf(Wk[i]); }
  if (i < Cd * Cd) { Wvb[i] = f2bf(Wv[i]); }
}

// ============================================================
// Kernel 1: xb[b][n][c] (bf16) = x[b][c][n]  — LDS tiled transpose
// ============================================================
__global__ __launch_bounds__(256)
void k_transpose_bf16(const float* __restrict__ x, unsigned short* __restrict__ xb) {
  __shared__ alignas(16) unsigned short tile[32][33];
  const int n0 = blockIdx.x * 32, c0 = blockIdx.y * 32, b = blockIdx.z;
  const int tx = threadIdx.x & 31, ty = threadIdx.x >> 5;   // 32 x 8
  #pragma unroll
  for (int yy = 0; yy < 4; ++yy) {
    const int c = c0 + ty + yy * 8;
    tile[ty + yy * 8][tx] = f2bf(x[((size_t)b * Cd + c) * Nd + n0 + tx]);
  }
  __syncthreads();
  #pragma unroll
  for (int yy = 0; yy < 4; ++yy) {
    const int n = n0 + ty + yy * 8;
    xb[((size_t)b * Nd + n) * Cd + c0 + tx] = tile[tx][ty + yy * 8];
  }
}

// ============================================================
// Kernel 2: Q/K projections.  Qb = (xb @ Wq^T + bq) * scale, Kb = xb @ Wk^T + bk
// One wave per 16-row n-tile, K=256 in 8 WMMA k-steps, 4 WMMAs per step.
// ============================================================
__global__ __launch_bounds__(128)
void k_proj_qk(const unsigned short* __restrict__ xb,
               const unsigned short* __restrict__ Wqb, const float* __restrict__ bq,
               const unsigned short* __restrict__ Wkb, const float* __restrict__ bk,
               unsigned short* __restrict__ Qb, unsigned short* __restrict__ Kb) {
  const int lane = threadIdx.x & 31;
  const int half = lane >> 4, lcol = lane & 15;
  const int wv = blockIdx.x * (blockDim.x >> 5) + (threadIdx.x >> 5);
  const int b  = wv >> 8;               // / (Nd/16)
  const int n0 = (wv & 255) * 16;

  v8f aq0 = zero8(), aq1 = zero8(), ak0 = zero8(), ak1 = zero8();
  const unsigned short* arow = xb + ((size_t)b * Nd + n0 + lcol) * Cd;

  for (int k0 = 0; k0 < Cd; k0 += 32) {
    const v16bf A   = load_row_operand(arow + k0, half);
    const v16bf Bq0 = load_row_operand(Wqb + (size_t)lcol        * Cd + k0, half);
    const v16bf Bq1 = load_row_operand(Wqb + (size_t)(16 + lcol) * Cd + k0, half);
    const v16bf Bk0 = load_row_operand(Wkb + (size_t)lcol        * Cd + k0, half);
    const v16bf Bk1 = load_row_operand(Wkb + (size_t)(16 + lcol) * Cd + k0, half);
    aq0 = wmma_bf16(A, Bq0, aq0);
    aq1 = wmma_bf16(A, Bq1, aq1);
    ak0 = wmma_bf16(A, Bk0, ak0);
    ak1 = wmma_bf16(A, Bk1, ak1);
  }

  const float scale = 1.0f / (sqrtf(32.0f) + 1e-8f);  // fold 1/(sqrt(d_k)+eps) into Q
  #pragma unroll
  for (int r = 0; r < 8; ++r) {
    const int n = n0 + r + half * 8;
    const size_t base = ((size_t)b * Nd + n) * Cq;
    Qb[base + lcol]      = f2bf((aq0[r] + bq[lcol])      * scale);
    Qb[base + 16 + lcol] = f2bf((aq1[r] + bq[16 + lcol]) * scale);
    Kb[base + lcol]      = f2bf(ak0[r] + bk[lcol]);
    Kb[base + 16 + lcol] = f2bf(ak1[r] + bk[16 + lcol]);
  }
}

// ============================================================
// Kernel 3: Vb[b][n][c] = xb @ Wv^T + bv   (bf16)
// One wave per (16-row n-tile, 64-col c-slab): 4 WMMA tiles, 8 k-steps.
// ============================================================
__global__ __launch_bounds__(128)
void k_proj_v(const unsigned short* __restrict__ xb,
              const unsigned short* __restrict__ Wvb, const float* __restrict__ bv,
              unsigned short* __restrict__ Vb) {
  const int lane = threadIdx.x & 31;
  const int half = lane >> 4, lcol = lane & 15;
  const int wv = blockIdx.x * (blockDim.x >> 5) + (threadIdx.x >> 5);
  const int b   = wv >> 10;             // / (256 n-tiles * 4 c-slabs)
  const int rem = wv & 1023;
  const int n0  = (rem >> 2) * 16;
  const int c0  = (rem & 3) * 64;

  v8f acc[4];
  #pragma unroll
  for (int t = 0; t < 4; ++t) acc[t] = zero8();

  const unsigned short* arow = xb + ((size_t)b * Nd + n0 + lcol) * Cd;
  for (int k0 = 0; k0 < Cd; k0 += 32) {
    const v16bf A = load_row_operand(arow + k0, half);
    #pragma unroll
    for (int t = 0; t < 4; ++t) {
      const v16bf Bv = load_row_operand(Wvb + (size_t)(c0 + t * 16 + lcol) * Cd + k0, half);
      acc[t] = wmma_bf16(A, Bv, acc[t]);
    }
  }
  #pragma unroll
  for (int t = 0; t < 4; ++t) {
    const int c = c0 + t * 16 + lcol;
    #pragma unroll
    for (int r = 0; r < 8; ++r) {
      const int n = n0 + r + half * 8;
      Vb[((size_t)b * Nd + n) * Cd + c] = f2bf(acc[t][r] + bv[c]);
    }
  }
}

// ============================================================
// Kernel 4: fused flash attention + epilogue.
//   energy clamped to +-5  =>  exp() is bounded, no online-max rescale needed.
// Block = 256 threads = 8 waves = 4 query tiles x 2 channel halves.
// Per 32-key step: async-stage K (2KB) into LDS while V^T (16KB) is staged
// with a transpose; S = Q K^T (2 WMMA), P = exp(clamp(S)), P bounced through
// per-wave LDS, O += P V^T (8 WMMA).
// ============================================================
__global__ __launch_bounds__(256)
void k_flash(const unsigned short* __restrict__ Qb,
             const unsigned short* __restrict__ Kb,
             const unsigned short* __restrict__ Vb,
             const float* __restrict__ gamma,
             const float* __restrict__ x,
             float* __restrict__ out) {
  __shared__ alignas(16) unsigned short Kl[32 * 32];        // [m][q]       2 KB
  __shared__ alignas(16) unsigned short Vt[Cd * 32];        // [c][m]      16 KB
  __shared__ alignas(16) unsigned short Pl[8][16 * 32];     // per-wave P   8 KB

  const int tid  = threadIdx.x;
  const int lane = tid & 31, wid = tid >> 5;
  const int half = lane >> 4, lcol = lane & 15;
  const int qt = wid >> 1;                      // query tile in block (0..3)
  const int ch = wid & 1;                       // channel half (0..1)
  const int b  = blockIdx.x >> 6;               // / (Nd/64)
  const int n0 = (blockIdx.x & 63) * 64 + qt * 16;
  const int c0 = ch * 128;

  // Q operand (16 x 32) lives in registers for the whole kernel.
  const v16bf Aq =
      load_row_operand(Qb + ((size_t)b * Nd + n0 + lcol) * Cq, half);

  v8f O[8];
  #pragma unroll
  for (int t = 0; t < 8; ++t) O[t] = zero8();
  float lsum[8];
  #pragma unroll
  for (int r = 0; r < 8; ++r) lsum[r] = 0.0f;

  const float gm = gamma[0];

  for (int m0 = 0; m0 < Nd; m0 += 32) {
    __syncthreads();                            // protect LDS reuse
    // ---- issue async K-tile copy: 32 rows x 64B = 128 x 16B chunks (waves 0-3)
    if (tid < 128) {
      const int mrow = tid >> 2, chunk = tid & 3;
      async_copy_b128(Kb + ((size_t)b * Nd + m0 + mrow) * Cq + chunk * 8,
                      Kl + mrow * 32 + chunk * 8);
    }
    // ---- overlap: stage V tile transposed, Vt[c][m], contiguous B-operand rows
    #pragma unroll
    for (int it = 0; it < 4; ++it) {
      const int idx  = tid + it * 256;          // 0..1023
      const int mrow = idx >> 5;
      const int cch  = idx & 31;
      const v8u vv = *(const v8u*)(Vb + ((size_t)b * Nd + m0 + mrow) * Cd + cch * 8);
      #pragma unroll
      for (int i = 0; i < 8; ++i) Vt[(cch * 8 + i) * 32 + mrow] = vv[i];
    }
    if (tid < 128) wait_async();                // drain ASYNCcnt before barrier
    __syncthreads();

    // ---- S = Q @ K^T : two 16x16 tiles (cols m0..15, m16..31)
    v8f S0 = zero8(), S1 = zero8();
    const v16bf Bk0 = load_row_operand(Kl + (lcol)      * 32, half);
    const v16bf Bk1 = load_row_operand(Kl + (16 + lcol) * 32, half);
    S0 = wmma_bf16(Aq, Bk0, S0);
    S1 = wmma_bf16(Aq, Bk1, S1);

    // ---- P = exp(clamp(S)); accumulate this lane's column partial row-sums;
    //      redistribute P (C-layout -> A-layout) through per-wave LDS scratch.
    unsigned short* pw = &Pl[wid][0];
    #pragma unroll
    for (int r = 0; r < 8; ++r) {
      const float p0 = __expf(fminf(fmaxf(S0[r], -CLAMP), CLAMP));
      const float p1 = __expf(fminf(fmaxf(S1[r], -CLAMP), CLAMP));
      lsum[r] += p0 + p1;
      const int row = r + half * 8;
      pw[row * 32 + lcol]      = f2bf(p0);
      pw[row * 32 + 16 + lcol] = f2bf(p1);
    }
    asm volatile("s_wait_dscnt 0x0" ::: "memory");   // P stores -> P loads (same wave)

    const v16bf Ap = load_row_operand(pw + lcol * 32, half);

    // ---- O += P @ V^T over this wave's 8 channel tiles
    #pragma unroll
    for (int t = 0; t < 8; ++t) {
      const v16bf Bv = load_row_operand(Vt + (size_t)(c0 + t * 16 + lcol) * 32, half);
      O[t] = wmma_bf16(Ap, Bv, O[t]);
    }

    // ---- prefetch next V tile into the cache hierarchy
    if (m0 + 32 < Nd) {
      __builtin_prefetch(Vb + ((size_t)b * Nd + m0 + 32 + lane) * Cd, 0, 1);
    }
  }

  // ---- finish row sums: reduce across the 16 lanes of each half-group
  #pragma unroll
  for (int r = 0; r < 8; ++r) {
    float v = lsum[r];
    v += __shfl_xor(v, 1, 32);
    v += __shfl_xor(v, 2, 32);
    v += __shfl_xor(v, 4, 32);
    v += __shfl_xor(v, 8, 32);
    lsum[r] = v;
  }

  // ---- epilogue: out[b][c][n] = gamma * O/l + x[b][c][n]
  #pragma unroll
  for (int t = 0; t < 8; ++t) {
    const int c = c0 + t * 16 + lcol;
    #pragma unroll
    for (int r = 0; r < 8; ++r) {
      const int n = n0 + r + half * 8;
      const size_t gi = ((size_t)b * Cd + c) * Nd + n;
      out[gi] = gm * (O[t][r] / lsum[r]) + x[gi];
    }
  }
}

// ============================================================
// launch
// ============================================================
extern "C" void kernel_launch(void* const* d_in, const int* in_sizes, int n_in,
                              void* d_out, int out_size, void* d_ws, size_t ws_size,
                              hipStream_t stream) {
  const float* x     = (const float*)d_in[0];
  const float* Wq    = (const float*)d_in[1];
  const float* bq    = (const float*)d_in[2];
  const float* Wk    = (const float*)d_in[3];
  const float* bk    = (const float*)d_in[4];
  const float* Wv    = (const float*)d_in[5];
  const float* bv    = (const float*)d_in[6];
  const float* gamma = (const float*)d_in[7];
  float* out = (float*)d_out;

  // workspace layout (bf16 = 2B):
  char* ws = (char*)d_ws;
  unsigned short* xb  = (unsigned short*)(ws);                                  //  8 MB
  unsigned short* Qb  = (unsigned short*)(ws + (size_t)Bn * Nd * Cd * 2);       //  1 MB
  unsigned short* Kb  = Qb + (size_t)Bn * Nd * Cq;                              //  1 MB
  unsigned short* Vb  = Kb + (size_t)Bn * Nd * Cq;                              //  8 MB
  unsigned short* Wqb = Vb + (size_t)Bn * Nd * Cd;                              // 16 KB
  unsigned short* Wkb = Wqb + (size_t)Cq * Cd;                                  // 16 KB
  unsigned short* Wvb = Wkb + (size_t)Cq * Cd;                                  // 128 KB

  k_cvt_weights<<<(Cd * Cd) / 256, 256, 0, stream>>>(Wq, Wk, Wv, Wqb, Wkb, Wvb);

  dim3 gT(Nd / 32, Cd / 32, Bn);
  k_transpose_bf16<<<gT, 256, 0, stream>>>(x, xb);

  // (Bn * Nd/16) waves / 4 waves per block = 256 blocks
  k_proj_qk<<<(Bn * (Nd / 16)) / 4, 128, 0, stream>>>(xb, Wqb, bq, Wkb, bk, Qb, Kb);

  // Bn * (Nd/16) * (Cd/64) = 4096 waves / 4 = 1024 blocks
  k_proj_v<<<(Bn * (Nd / 16) * (Cd / 64)) / 4, 128, 0, stream>>>(xb, Wvb, bv, Vb);

  // Bn * Nd/64 = 256 blocks of 8 waves
  k_flash<<<Bn * (Nd / 64), 256, 0, stream>>>(Qb, Kb, Vb, gamma, x, out);
}
